// Model_59785944760477
// MI455X (gfx1250) — matrix operand order repaired and checked by reading, not compiled, not run on hardware
//
#include <hip/hip_runtime.h>
#include <hip/hip_bf16.h>

// ---------------------------------------------------------------------------
// Model constants (from the reference)
// ---------------------------------------------------------------------------
#define HID 128
#define NC 100000
#define NA 1024
#define FC 256
#define FA 4096

// ---------------------------------------------------------------------------
// vector typedefs
// ---------------------------------------------------------------------------
typedef __attribute__((ext_vector_type(16))) __bf16 v16bf;
typedef __attribute__((ext_vector_type(8)))  __bf16 v8bf;
typedef __attribute__((ext_vector_type(8)))  float  v8f;
typedef __attribute__((ext_vector_type(4)))  unsigned int v4u;
typedef __attribute__((ext_vector_type(8)))  int v8i_t;
typedef __attribute__((ext_vector_type(4)))  int v4i_t;

static __device__ __forceinline__ unsigned short f2bfu(float f) {
    unsigned u = __float_as_uint(f);
    u += 0x7FFFu + ((u >> 16) & 1u);   // round-to-nearest-even
    return (unsigned short)(u >> 16);
}
static __device__ __forceinline__ float bfu2f(unsigned short s) {
    return __uint_as_float(((unsigned)s) << 16);
}

// ---------------------------------------------------------------------------
// fp32 -> bf16 conversion
// ---------------------------------------------------------------------------
__global__ void k_f2bf(const float* __restrict__ in, unsigned short* __restrict__ out, int n) {
    int i = blockIdx.x * blockDim.x + threadIdx.x;
    if (i < n) out[i] = f2bfu(in[i]);
}

// Split Wd1 [128,256] into A = Wd1[:, :128], B = Wd1[:, 128:] (both bf16 [128,128])
__global__ void k_pack_wd1(const float* __restrict__ Wd1,
                           unsigned short* __restrict__ A, unsigned short* __restrict__ B) {
    int i = blockIdx.x * blockDim.x + threadIdx.x;
    if (i >= 128 * 128) return;
    int n = i >> 7, k = i & 127;
    A[i] = f2bfu(Wd1[n * 256 + k]);
    B[i] = f2bfu(Wd1[n * 256 + 128 + k]);
}

// ---------------------------------------------------------------------------
// Stage 0: artval[j] = article_x . W_lin1[j,:] + b_lin1[j]   (1024 outputs)
// ---------------------------------------------------------------------------
__global__ void k_art_gemv(const float* __restrict__ ax, const float* __restrict__ W,
                           const float* __restrict__ b, float* __restrict__ artval) {
    int j = blockIdx.x * blockDim.x + threadIdx.x;
    if (j >= NA) return;
    const float4* wr = (const float4*)(W + (size_t)j * FA);
    const float4* av = (const float4*)ax;
    float s = 0.f;
    for (int k = 0; k < FA / 4; ++k) {
        float4 w = wr[k], a = av[k];
        s += w.x * a.x + w.y * a.y + w.z * a.z + w.w * a.w;
    }
    artval[j] = s + b[j];
}

// ---------------------------------------------------------------------------
// Scalar edge aggregation: agg[dst] += artval[src]; cnt[dst] += 1
// ei layout [2, E] row-major: src at [e], dst at [E + e]
// ---------------------------------------------------------------------------
__global__ void k_edge_scalar_agg(const int* __restrict__ ei, int E,
                                  const float* __restrict__ artval,
                                  float* __restrict__ agg, float* __restrict__ cnt) {
    int e = blockIdx.x * blockDim.x + threadIdx.x;
    if (e >= E) return;
    int s = ei[e], d = ei[E + e];
    atomicAdd(&agg[d], artval[s]);
    atomicAdd(&cnt[d], 1.0f);
}

__global__ void k_mean_scalar(const float* __restrict__ agg, const float* __restrict__ cnt,
                              float* __restrict__ mean, int n) {
    int i = blockIdx.x * blockDim.x + threadIdx.x;
    if (i < n) mean[i] = agg[i] / fmaxf(cnt[i], 1.0f);
}

// ---------------------------------------------------------------------------
// Vector edge aggregation (e_cc): one wave32 per edge, 128 features
// ---------------------------------------------------------------------------
__global__ __launch_bounds__(256)
void k_edge_vec_agg(const int* __restrict__ ei, int E, const unsigned short* __restrict__ h2,
                    float* __restrict__ agg, float* __restrict__ cnt) {
    int wave = (blockIdx.x * blockDim.x + threadIdx.x) >> 5;
    int lane = threadIdx.x & 31;
    if (wave >= E) return;
    int s = ei[wave], d = ei[E + wave];
    size_t sb = (size_t)s * HID, db = (size_t)d * HID;
#pragma unroll
    for (int j = 0; j < 4; ++j) {
        int f = lane + 32 * j;
        atomicAdd(&agg[db + f], bfu2f(h2[sb + f]));
    }
    if (lane == 0) atomicAdd(&cnt[d], 1.0f);
}

__global__ void k_mean_vec_bf(const float* __restrict__ agg, const float* __restrict__ cnt,
                              unsigned short* __restrict__ out, int n) {
    int i = blockIdx.x * blockDim.x + threadIdx.x;
    if (i >= n) return;
    int node = i >> 7;
    out[i] = f2bfu(agg[i] / fmaxf(cnt[node], 1.0f));
}

// ---------------------------------------------------------------------------
// WMMA GEMM (templated on K): out[m,n] = act( X[M,K] @ W[128,K].T  + epilogue )
//
// Weights are staged into LDS once per block by the Tensor Data Mover:
//   2D descriptor, tile = whole [128 x K] bf16 matrix, with pad_enable so that
//   4 DWORDs (16B) of padding are inserted after each row (K*2 bytes of data):
//   LDS row stride = (K+8) bf16 elements -> B-fragment ds_load_b128 reads are
//   16B-aligned and bank-conflict-free (lane n -> banks 4n..4n+3).
//
// Fragments (v_wmma_f32_16x16x32_bf16):
//   A 16x32: lane(m=lane&15, half=lane>>4): elems 0..7  = K kt*32+half*8+{0..7}
//                                           elems 8..15 = +16
//   B 32x16: lane(n=nt*16+(lane&15), kh=lane>>4): same K order as A (kt*32+kh*8+{0..7}, +16)
//   C/D 16x16 f32: col n=lane&15 (+nt*16), rows m=(lane>>4)*8 + r
// ---------------------------------------------------------------------------
template<int K>
__global__ __launch_bounds__(256)
void k_gemm_bf16_n128(const unsigned short* __restrict__ X, const unsigned short* __restrict__ Wt,
                      int M,
                      const float* __restrict__ scaleVec, const float* __restrict__ wlcol,
                      const float* __restrict__ bias, const float* __restrict__ cin,
                      unsigned short* __restrict__ outBf, float* __restrict__ outF, int relu) {
    extern __shared__ unsigned short Wlds[];           // 128 * (K+8) bf16
    constexpr int LSTRIDE = K + 8;                     // elements
    constexpr int NKT = K / 32;

    const int lane = threadIdx.x & 31;
    const int wave = threadIdx.x >> 5;
    const int rowBase = (blockIdx.x * 8 + wave) * 16;
    const bool active = rowBase < M;

    // ---- TDM: stage W[128,K] bf16 into LDS with 16B per-row padding ----
    if (threadIdx.x < 32) {
        unsigned long long ga = (unsigned long long)Wt;
        unsigned lds_addr = (unsigned)(unsigned long long)&Wlds[0];
        constexpr unsigned padIntervalCode = (K == 128) ? 5u : 6u;  // 64 / 128 DWORDs of data
        constexpr unsigned padAmountCode   = 3u;                    // 4 DWORDs = 16B
        v4u g0 = { 1u,                                  // count=1, user descriptor
                   lds_addr,
                   (unsigned)(ga & 0xFFFFFFFFull),
                   (unsigned)((ga >> 32) & 0x01FFFFFFull) | (2u << 30) }; // type=2 (image)
        v8i_t g1 = { (int)((1u << 16) | (1u << 20) | (padIntervalCode << 22) | (padAmountCode << 25)),
                     (int)(((unsigned)K & 0xFFFFu) << 16),          // tensor_dim0 lo
                     (int)((128u & 0xFFFFu) << 16),                 // td0 hi | tensor_dim1 lo
                     (int)(((unsigned)K) << 16),                    // td1 hi | tile_dim0
                     (int)(128u),                                   // tile_dim1 | tile_dim2=0
                     (int)K,                                        // tensor_dim0_stride lo
                     0, 0 };                                        // strides hi / dim1_stride
        v4i_t gz4 = { 0, 0, 0, 0 };
        v8i_t gz8 = { 0, 0, 0, 0, 0, 0, 0, 0 };
        __builtin_amdgcn_tensor_load_to_lds(g0, g1, gz4, gz4, gz8, 0);
        __builtin_amdgcn_s_wait_tensorcnt(0);
    }
    __syncthreads();

    if (active) {
        const int half = lane >> 4;
        const int mfrag = lane & 15;

        v8f acc[8];
#pragma unroll
        for (int i = 0; i < 8; ++i) acc[i] = (v8f){0.f, 0.f, 0.f, 0.f, 0.f, 0.f, 0.f, 0.f};

        const unsigned short* xrow = X + (size_t)(rowBase + mfrag) * K;
        const int nbase = lane & 15;

#pragma unroll
        for (int kt = 0; kt < NKT; ++kt) {
            // A fragment: two contiguous 16B chunks from global (streamed)
            const __bf16* ap = (const __bf16*)(xrow + kt * 32);
            v8bf alo = *(const v8bf*)(ap + half * 8);
            v8bf ahi = *(const v8bf*)(ap + 16 + half * 8);
            v16bf a;
#pragma unroll
            for (int i = 0; i < 8; ++i) { a[i] = alo[i]; a[i + 8] = ahi[i]; }

            // B fragments: batch-load all 8 from LDS, then 8 back-to-back WMMAs
            v16bf bfrag[8];
#pragma unroll
            for (int nt = 0; nt < 8; ++nt) {
                const int n = nt * 16 + nbase;
                const __bf16* bp = (const __bf16*)(Wlds + (size_t)n * LSTRIDE + kt * 32 + half * 8);
                v8bf blo = *(const v8bf*)bp;
                v8bf bhi = *(const v8bf*)(bp + 16);
#pragma unroll
                for (int i = 0; i < 8; ++i) { bfrag[nt][i] = blo[i]; bfrag[nt][i + 8] = bhi[i]; }
            }
#pragma unroll
            for (int nt = 0; nt < 8; ++nt) {
                acc[nt] = __builtin_amdgcn_wmma_f32_16x16x32_bf16(
                    false, a, false, bfrag[nt], (short)0, acc[nt], false, false);
            }
        }

        // ---- epilogue ----
        const int mrow0 = rowBase + (lane >> 4) * 8;
        float sv[8];
        if (scaleVec) {
#pragma unroll
            for (int r = 0; r < 8; ++r) sv[r] = scaleVec[mrow0 + r];
        }
#pragma unroll
        for (int nt = 0; nt < 8; ++nt) {
            int n = nt * 16 + nbase;
            float wl = wlcol ? wlcol[n] : 0.0f;
            float bs = bias ? bias[n] : 0.0f;
#pragma unroll
            for (int r = 0; r < 8; ++r) {
                size_t idx = (size_t)(mrow0 + r) * 128 + n;
                float v = acc[nt][r] + bs;
                if (scaleVec) v += sv[r] * wl;
                if (cin) v += cin[idx];
                if (relu) v = fmaxf(v, 0.0f);
                if (outF) outF[idx] = v;
                if (outBf) outBf[idx] = f2bfu(v);
            }
        }
    }
}

// ---------------------------------------------------------------------------
// Edge decoder: out[e] = Wd2 . relu(P[row] + Q[col] + bd1) + bd2
// ---------------------------------------------------------------------------
__global__ __launch_bounds__(256)
void k_edge_decoder(const int* __restrict__ ei, int E,
                    const float* __restrict__ P, const float* __restrict__ Q,
                    const float* __restrict__ bd1, const float* __restrict__ Wd2,
                    const float* __restrict__ bd2, float* __restrict__ out) {
    int wave = (blockIdx.x * blockDim.x + threadIdx.x) >> 5;
    int lane = threadIdx.x & 31;
    if (wave >= E) return;
    int r = ei[wave], c = ei[E + wave];
    float s = 0.f;
#pragma unroll
    for (int j = 0; j < 4; ++j) {
        int f = lane + 32 * j;
        float u = P[(size_t)r * 128 + f] + Q[(size_t)c * 128 + f] + bd1[f];
        s += fmaxf(u, 0.0f) * Wd2[f];
    }
#pragma unroll
    for (int off = 16; off; off >>= 1) s += __shfl_xor(s, off, 32);
    if (lane == 0) out[wave] = s + bd2[0];
}

// ---------------------------------------------------------------------------
// Host launcher
// ---------------------------------------------------------------------------
extern "C" void kernel_launch(void* const* d_in, const int* in_sizes, int n_in,
                              void* d_out, int out_size, void* d_ws, size_t ws_size,
                              hipStream_t stream) {
    const float* article_x  = (const float*)d_in[0];
    const float* community_x= (const float*)d_in[1];
    const int*   e_wb       = (const int*)d_in[2];
    const int*   e_mb       = (const int*)d_in[3];
    const int*   e_cc       = (const int*)d_in[4];
    const float* W_lin1     = (const float*)d_in[5];
    const float* b_lin1     = (const float*)d_in[6];
    const float* Wl1        = (const float*)d_in[7];
    const float* bl1        = (const float*)d_in[8];
    const float* Wr1        = (const float*)d_in[9];
    const float* Wl2        = (const float*)d_in[10];
    const float* bl2        = (const float*)d_in[11];
    const float* Wr2        = (const float*)d_in[12];
    const float* Wl3        = (const float*)d_in[13];
    const float* bl3        = (const float*)d_in[14];
    const float* Wr3        = (const float*)d_in[15];
    const float* W_lin2     = (const float*)d_in[16];
    const float* b_lin2     = (const float*)d_in[17];
    const float* Wd1        = (const float*)d_in[18];
    const float* bd1        = (const float*)d_in[19];
    const float* Wd2        = (const float*)d_in[20];
    const float* bd2        = (const float*)d_in[21];
    float* out = (float*)d_out;

    const int E_wb = in_sizes[2] / 2;
    const int E_mb = in_sizes[3] / 2;
    const int E_cc = in_sizes[4] / 2;

    // --- workspace bump allocator (256B aligned) ---
    char* base = (char*)d_ws;
    size_t off = 0;
    auto alloc = [&](size_t bytes) -> void* {
        void* p = base + off;
        off = (off + bytes + 255) & ~(size_t)255;
        return p;
    };
    unsigned short* cx_bf   = (unsigned short*)alloc((size_t)NC * FC * 2);   // reused as P later
    unsigned short* h1_bf   = (unsigned short*)alloc((size_t)NC * HID * 2);
    unsigned short* h2_bf   = (unsigned short*)alloc((size_t)NC * HID * 2);
    unsigned short* h3_bf   = (unsigned short*)alloc((size_t)NC * HID * 2);
    unsigned short* z_bf    = (unsigned short*)alloc((size_t)NC * HID * 2);
    unsigned short* m3_bf   = (unsigned short*)alloc((size_t)NC * HID * 2);
    float* agg3             = (float*)alloc((size_t)NC * HID * 4);           // reused as tmp
    float* Qbuf             = (float*)alloc((size_t)NC * HID * 4);
    float* artval           = (float*)alloc((size_t)NA * 4);
    float* agg1             = (float*)alloc((size_t)NC * 4);
    float* cnt1             = (float*)alloc((size_t)NC * 4);
    float* agg2             = (float*)alloc((size_t)NC * 4);
    float* cnt2             = (float*)alloc((size_t)NC * 4);
    float* cnt3             = (float*)alloc((size_t)NC * 4);
    float* mean1            = (float*)alloc((size_t)NC * 4);
    float* mean2            = (float*)alloc((size_t)NC * 4);
    unsigned short* wr1_bf  = (unsigned short*)alloc((size_t)HID * FC * 2);
    unsigned short* wr2_bf  = (unsigned short*)alloc((size_t)HID * HID * 2);
    unsigned short* wl3_bf  = (unsigned short*)alloc((size_t)HID * HID * 2);
    unsigned short* wr3_bf  = (unsigned short*)alloc((size_t)HID * FC * 2);
    unsigned short* wz_bf   = (unsigned short*)alloc((size_t)HID * HID * 2);
    unsigned short* wdA_bf  = (unsigned short*)alloc((size_t)HID * HID * 2);
    unsigned short* wdB_bf  = (unsigned short*)alloc((size_t)HID * HID * 2);
    float* Pbuf  = (float*)cx_bf;   // alias: cx_bf dead after layer-3 GEMM
    float* tmp3  = agg3;            // alias: agg3 dead after mean3

    (void)hipMemsetAsync(agg1, 0, (size_t)NC * 4, stream);
    (void)hipMemsetAsync(cnt1, 0, (size_t)NC * 4, stream);
    (void)hipMemsetAsync(agg2, 0, (size_t)NC * 4, stream);
    (void)hipMemsetAsync(cnt2, 0, (size_t)NC * 4, stream);
    (void)hipMemsetAsync(cnt3, 0, (size_t)NC * 4, stream);
    (void)hipMemsetAsync(agg3, 0, (size_t)NC * HID * 4, stream);

    const int BT = 256;
    auto nb = [](long long n, int b) { return (unsigned)((n + b - 1) / b); };

    // --- conversions ---
    k_f2bf<<<nb((long long)NC * FC, BT), BT, 0, stream>>>(community_x, cx_bf, NC * FC);
    k_f2bf<<<nb(HID * FC, BT), BT, 0, stream>>>(Wr1, wr1_bf, HID * FC);
    k_f2bf<<<nb(HID * HID, BT), BT, 0, stream>>>(Wr2, wr2_bf, HID * HID);
    k_f2bf<<<nb(HID * HID, BT), BT, 0, stream>>>(Wl3, wl3_bf, HID * HID);
    k_f2bf<<<nb(HID * FC, BT), BT, 0, stream>>>(Wr3, wr3_bf, HID * FC);
    k_f2bf<<<nb(HID * HID, BT), BT, 0, stream>>>(W_lin2, wz_bf, HID * HID);
    k_pack_wd1<<<nb(HID * HID, BT), BT, 0, stream>>>(Wd1, wdA_bf, wdB_bf);

    // --- stage 0 + scalar segment means ---
    k_art_gemv<<<nb(NA, BT), BT, 0, stream>>>(article_x, W_lin1, b_lin1, artval);
    k_edge_scalar_agg<<<nb(E_wb, BT), BT, 0, stream>>>(e_wb, E_wb, artval, agg1, cnt1);
    k_edge_scalar_agg<<<nb(E_mb, BT), BT, 0, stream>>>(e_mb, E_mb, artval, agg2, cnt2);
    k_mean_scalar<<<nb(NC, BT), BT, 0, stream>>>(agg1, cnt1, mean1, NC);
    k_mean_scalar<<<nb(NC, BT), BT, 0, stream>>>(agg2, cnt2, mean2, NC);

    const unsigned gemmGrid = nb(NC / 16, 8);                 // 6250 tiles / 8 waves
    const size_t lds128 = (size_t)128 * (128 + 8) * 2;        // 34,816 B
    const size_t lds256 = (size_t)128 * (256 + 8) * 2;        // 67,584 B

    // --- layer 1: h1 = relu(mean1*Wl1col + bl1 + cx @ Wr1.T) ---
    k_gemm_bf16_n128<256><<<gemmGrid, BT, lds256, stream>>>(cx_bf, wr1_bf, NC,
                                                            mean1, Wl1, bl1, nullptr,
                                                            h1_bf, nullptr, 1);
    // --- layer 2: h2 = relu(mean2*Wl2col + bl2 + h1 @ Wr2.T) ---
    k_gemm_bf16_n128<128><<<gemmGrid, BT, lds128, stream>>>(h1_bf, wr2_bf, NC,
                                                            mean2, Wl2, bl2, nullptr,
                                                            h2_bf, nullptr, 1);

    // --- layer-3 aggregation over e_cc ---
    k_edge_vec_agg<<<nb((long long)E_cc * 32, BT), BT, 0, stream>>>(e_cc, E_cc, h2_bf, agg3, cnt3);
    k_mean_vec_bf<<<nb((long long)NC * HID, BT), BT, 0, stream>>>(agg3, cnt3, m3_bf, NC * HID);

    // --- layer 3: tmp = mean3 @ Wl3.T ; h3 = relu(tmp + bl3 + cx @ Wr3.T) ---
    k_gemm_bf16_n128<128><<<gemmGrid, BT, lds128, stream>>>(m3_bf, wl3_bf, NC,
                                                            nullptr, nullptr, nullptr, nullptr,
                                                            nullptr, tmp3, 0);
    k_gemm_bf16_n128<256><<<gemmGrid, BT, lds256, stream>>>(cx_bf, wr3_bf, NC,
                                                            nullptr, nullptr, bl3, tmp3,
                                                            h3_bf, nullptr, 1);

    // --- z = h3 @ W_lin2.T + b_lin2 ---
    k_gemm_bf16_n128<128><<<gemmGrid, BT, lds128, stream>>>(h3_bf, wz_bf, NC,
                                                            nullptr, nullptr, b_lin2, nullptr,
                                                            z_bf, nullptr, 0);

    // --- decoder factorization: P = z @ Wd1[:,:128].T ; Q = z @ Wd1[:,128:].T ---
    k_gemm_bf16_n128<128><<<gemmGrid, BT, lds128, stream>>>(z_bf, wdA_bf, NC,
                                                            nullptr, nullptr, nullptr, nullptr,
                                                            nullptr, Pbuf, 0);
    k_gemm_bf16_n128<128><<<gemmGrid, BT, lds128, stream>>>(z_bf, wdB_bf, NC,
                                                            nullptr, nullptr, nullptr, nullptr,
                                                            nullptr, Qbuf, 0);

    // --- per-edge decode ---
    k_edge_decoder<<<nb((long long)E_cc * 32, BT), BT, 0, stream>>>(e_cc, E_cc, Pbuf, Qbuf,
                                                                    bd1, Wd2, bd2, out);
    (void)n_in; (void)out_size; (void)ws_size;
}
